// DeepseekV2MoEBlock_31310311587978
// MI455X (gfx1250) — matrix operand order repaired and checked by reading, not compiled, not run on hardware
//
#include <hip/hip_runtime.h>

// ---- model dims --------------------------------------------------------
constexpr int HD  = 2048;   // hidden size
constexpr int ID  = 1408;   // moe intermediate
constexpr int NE  = 16;     // routed experts
constexpr int ISD = 2816;   // shared intermediate (I * n_shared)
constexpr int NT  = 2048;   // tokens (B*S)
constexpr int CAP = 1024;   // expert capacity

// ---- vector types for WMMA (gfx1250, wave32) ---------------------------
typedef __attribute__((ext_vector_type(16))) __bf16         v16bf;
typedef __attribute__((ext_vector_type(8)))  float          v8f;
typedef __attribute__((ext_vector_type(16))) unsigned short v16u;
typedef __attribute__((ext_vector_type(8)))  unsigned short v8u;

__device__ __forceinline__ unsigned short f2bf(float f) {
    union { float f; unsigned u; } c; c.f = f;
    unsigned r = c.u + 0x7FFFu + ((c.u >> 16) & 1u);   // round-to-nearest-even
    return (unsigned short)(r >> 16);
}

__device__ __forceinline__ v16bf as_bf(v16u a) {
    union { v16u u; v16bf b; } c; c.u = a; return c.b;
}

// A-fragment (16x32 bf16 tile of a row-major [M][K] matrix): lane holds row
// M = lane%16; element e -> K = k0 + laneHi*16 + e.
__device__ __forceinline__ v16bf load_a_frag(const unsigned short* rowk, int laneHi) {
    v8u lo = *(const v8u*)(rowk + laneHi * 16);
    v8u hi = *(const v8u*)(rowk + laneHi * 16 + 8);
    v16u a;
#pragma unroll
    for (int i = 0; i < 8; i++) { a[i] = lo[i]; a[i + 8] = hi[i]; }
    return as_bf(a);
}

// B-fragment (32x16) from a PRE-TRANSPOSED bf16 weight WT[N][K] (row-major,
// leading dim ldk): lane holds column N, element e -> K = k0 + laneHi*16 + e,
// i.e. 16 contiguous bf16 = one aligned 32-byte load.
__device__ __forceinline__ v16bf load_bt_frag(const unsigned short* __restrict__ WT,
                                              size_t ldk, int n, int k0, int laneHi) {
    return as_bf(*(const v16u*)(WT + (size_t)n * ldk + k0 + laneHi * 16));
}

__device__ __forceinline__ v8f wmma_bf16(v16bf a, v16bf b, v8f c) {
    return __builtin_amdgcn_wmma_f32_16x16x32_bf16(
        /*neg_a=*/false, a, /*neg_b=*/false, b,
        /*c_mod=*/(short)0, c, /*reuse_a=*/false, /*reuse_b=*/false);
}

__device__ __forceinline__ float silu(float g) { return g / (1.f + __expf(-g)); }

// ---- kernel 0: tiled fp32 -> bf16 transpose ([R][C] -> [C][R]) ---------
__global__ __launch_bounds__(256) void k_transpose_cvt(const float* __restrict__ src,
                                                       unsigned short* __restrict__ dst,
                                                       int R, int C,
                                                       size_t src_batch, size_t dst_batch) {
    __shared__ unsigned short tile[32][33];
    const float* s = src + (size_t)blockIdx.z * src_batch;
    unsigned short* d = dst + (size_t)blockIdx.z * dst_batch;
    const int c0 = blockIdx.x * 32, r0 = blockIdx.y * 32;
    const int tx = threadIdx.x & 31, ty = threadIdx.x >> 5;   // 8 rows per pass
#pragma unroll
    for (int i = 0; i < 32; i += 8)
        tile[ty + i][tx] = f2bf(s[(size_t)(r0 + ty + i) * C + c0 + tx]);
    __syncthreads();
#pragma unroll
    for (int i = 0; i < 32; i += 8)
        d[(size_t)(c0 + ty + i) * R + r0 + tx] = tile[tx][ty + i];
}

// ---- kernel 1: fp32 x -> bf16 ------------------------------------------
__global__ __launch_bounds__(256) void k_cvt_x(const float* __restrict__ x,
                                               unsigned short* __restrict__ xb) {
    int i = (blockIdx.x * 256 + threadIdx.x) * 4;
    float4 v = *(const float4*)(x + i);
    ushort4 o;
    o.x = f2bf(v.x); o.y = f2bf(v.y); o.z = f2bf(v.z); o.w = f2bf(v.w);
    *(ushort4*)(xb + i) = o;
}

// ---- kernel 2: router (softmax over 16 logits, top-2, dispatch) --------
__global__ __launch_bounds__(32) void k_router(const float* __restrict__ x,
                                               const float* __restrict__ gate_w,
                                               int* __restrict__ counts,
                                               int* __restrict__ slot_token,
                                               int* __restrict__ slot_dst,
                                               float* __restrict__ slot_wt) {
    const int t = blockIdx.x;
    const int lane = threadIdx.x;
    const float* xr = x + (size_t)t * HD;
    float p[NE];
#pragma unroll
    for (int e = 0; e < NE; e++) p[e] = 0.f;
    for (int h = lane; h < HD; h += 32) {
        float xv = xr[h];
#pragma unroll
        for (int e = 0; e < NE; e++) p[e] += xv * gate_w[e * HD + h];
    }
#pragma unroll
    for (int e = 0; e < NE; e++) {
        float v = p[e];
#pragma unroll
        for (int o = 16; o > 0; o >>= 1) v += __shfl_xor(v, o, 32);
        p[e] = v;
    }
    if (lane == 0) {
        float mx = p[0];
#pragma unroll
        for (int e = 1; e < NE; e++) mx = fmaxf(mx, p[e]);
        float sum = 0.f;
#pragma unroll
        for (int e = 0; e < NE; e++) { p[e] = __expf(p[e] - mx); sum += p[e]; }
        float inv = 1.f / sum;
#pragma unroll
        for (int e = 0; e < NE; e++) p[e] *= inv;
        int i0 = 0; float v0 = p[0];
#pragma unroll
        for (int e = 1; e < NE; e++) if (p[e] > v0) { v0 = p[e]; i0 = e; }
        int i1 = -1; float v1 = -1.f;
#pragma unroll
        for (int e = 0; e < NE; e++) if (e != i0 && p[e] > v1) { v1 = p[e]; i1 = e; }
        int pos0 = atomicAdd(&counts[i0], 1);
        if (pos0 < CAP) {
            slot_token[i0 * CAP + pos0] = t;
            slot_dst[i0 * CAP + pos0]   = t * 2 + 0;
            slot_wt[i0 * CAP + pos0]    = v0;
        }
        int pos1 = atomicAdd(&counts[i1], 1);
        if (pos1 < CAP) {
            slot_token[i1 * CAP + pos1] = t;
            slot_dst[i1 * CAP + pos1]   = t * 2 + 1;
            slot_wt[i1 * CAP + pos1]    = v1;
        }
    }
}

// ---- kernel 3: routed gate/up proj + SwiGLU -> bf16 hbuf ---------------
// 32(M) x 64(N) per wave: each B fragment feeds 2 WMMAs (A-tile reuse).
// wtg/wtu are transposed bf16: [E][ID][HD]
__global__ __launch_bounds__(32) void k_gateup(const unsigned short* __restrict__ wtg,
                                               const unsigned short* __restrict__ wtu,
                                               const unsigned short* __restrict__ xb,
                                               const int* __restrict__ counts,
                                               const int* __restrict__ slot_token,
                                               unsigned short* __restrict__ hbuf) {
    const int e  = blockIdx.z;
    const int m0 = blockIdx.y * 32;
    const int n0 = blockIdx.x * 64;
    const int cnt = counts[e];
    if (m0 >= cnt) return;
    const int lane = threadIdx.x, laneHi = lane >> 4, lnLo = lane & 15;
    const int row0 = m0 + lnLo, row1 = m0 + 16 + lnLo;
    const int tok0 = (row0 < cnt) ? slot_token[e * CAP + row0] : 0;
    const int tok1 = (row1 < cnt) ? slot_token[e * CAP + row1] : 0;
    const unsigned short* arow0 = xb + (size_t)tok0 * HD;
    const unsigned short* arow1 = xb + (size_t)tok1 * HD;
    const unsigned short* Wg = wtg + (size_t)e * ID * HD;   // [ID][HD]
    const unsigned short* Wu = wtu + (size_t)e * ID * HD;
    v8f accg[2][4], accu[2][4];
#pragma unroll
    for (int s = 0; s < 2; s++)
#pragma unroll
        for (int j = 0; j < 4; j++) {
            accg[s][j] = v8f{0.f,0.f,0.f,0.f,0.f,0.f,0.f,0.f};
            accu[s][j] = v8f{0.f,0.f,0.f,0.f,0.f,0.f,0.f,0.f};
        }
    for (int k0 = 0; k0 < HD; k0 += 32) {
        __builtin_prefetch(Wg + (size_t)(n0 + lnLo) * HD + k0 + 256, 0, 3);
        __builtin_prefetch(Wu + (size_t)(n0 + lnLo) * HD + k0 + 256, 0, 3);
        v16bf a0 = load_a_frag(arow0 + k0, laneHi);
        v16bf a1 = load_a_frag(arow1 + k0, laneHi);
#pragma unroll
        for (int j = 0; j < 4; j++) {
            v16bf b = load_bt_frag(Wg, HD, n0 + 16 * j + lnLo, k0, laneHi);
            accg[0][j] = wmma_bf16(a0, b, accg[0][j]);
            accg[1][j] = wmma_bf16(a1, b, accg[1][j]);
        }
#pragma unroll
        for (int j = 0; j < 4; j++) {
            v16bf b = load_bt_frag(Wu, HD, n0 + 16 * j + lnLo, k0, laneHi);
            accu[0][j] = wmma_bf16(a0, b, accu[0][j]);
            accu[1][j] = wmma_bf16(a1, b, accu[1][j]);
        }
    }
#pragma unroll
    for (int s = 0; s < 2; s++)
#pragma unroll
        for (int j = 0; j < 4; j++) {
            const int n = n0 + 16 * j + lnLo;
#pragma unroll
            for (int r = 0; r < 8; r++) {
                const int m = m0 + s * 16 + laneHi * 8 + r;
                float h = silu(accg[s][j][r]) * accu[s][j][r];
                hbuf[((size_t)e * CAP + m) * ID + n] = f2bf(h);
            }
        }
}

// ---- kernel 4: routed down proj, scaled by router weight -> out2 -------
// wtd is transposed bf16: [E][HD][ID]
__global__ __launch_bounds__(32) void k_down(const unsigned short* __restrict__ wtd,
                                             const unsigned short* __restrict__ hbuf,
                                             const int* __restrict__ counts,
                                             const int* __restrict__ slot_dst,
                                             const float* __restrict__ slot_wt,
                                             float* __restrict__ out2) {
    const int e  = blockIdx.z;
    const int m0 = blockIdx.y * 32;
    const int n0 = blockIdx.x * 64;
    const int cnt = counts[e];
    if (m0 >= cnt) return;
    const int lane = threadIdx.x, laneHi = lane >> 4, lnLo = lane & 15;
    const unsigned short* arow0 = hbuf + ((size_t)e * CAP + (m0 + lnLo)) * ID;
    const unsigned short* arow1 = hbuf + ((size_t)e * CAP + (m0 + 16 + lnLo)) * ID;
    const unsigned short* Wd = wtd + (size_t)e * HD * ID;   // [HD][ID]
    v8f acc[2][4];
#pragma unroll
    for (int s = 0; s < 2; s++)
#pragma unroll
        for (int j = 0; j < 4; j++) acc[s][j] = v8f{0.f,0.f,0.f,0.f,0.f,0.f,0.f,0.f};
    for (int k0 = 0; k0 < ID; k0 += 32) {
        __builtin_prefetch(Wd + (size_t)(n0 + lnLo) * ID + k0 + 256, 0, 3);
        v16bf a0 = load_a_frag(arow0 + k0, laneHi);
        v16bf a1 = load_a_frag(arow1 + k0, laneHi);
#pragma unroll
        for (int j = 0; j < 4; j++) {
            v16bf b = load_bt_frag(Wd, ID, n0 + 16 * j + lnLo, k0, laneHi);
            acc[0][j] = wmma_bf16(a0, b, acc[0][j]);
            acc[1][j] = wmma_bf16(a1, b, acc[1][j]);
        }
    }
#pragma unroll
    for (int s = 0; s < 2; s++)
#pragma unroll
        for (int j = 0; j < 4; j++) {
            const int n = n0 + 16 * j + lnLo;
#pragma unroll
            for (int r = 0; r < 8; r++) {
                const int m = m0 + s * 16 + laneHi * 8 + r;
                if (m < cnt) {
                    const int   dst = slot_dst[e * CAP + m];
                    const float w   = slot_wt[e * CAP + m];
                    out2[(size_t)dst * HD + n] = w * acc[s][j][r];
                }
            }
        }
}

// ---- kernel 5: shared gate/up + SwiGLU -> bf16 sbuf --------------------
// wsgT/wsuT: [ISD][HD] bf16
__global__ __launch_bounds__(32) void k_shared_gateup(const unsigned short* __restrict__ wsgT,
                                                      const unsigned short* __restrict__ wsuT,
                                                      const unsigned short* __restrict__ xb,
                                                      unsigned short* __restrict__ sbuf) {
    const int t0 = blockIdx.y * 32;
    const int n0 = blockIdx.x * 64;
    const int lane = threadIdx.x, laneHi = lane >> 4, lnLo = lane & 15;
    const unsigned short* arow0 = xb + (size_t)(t0 + lnLo) * HD;
    const unsigned short* arow1 = xb + (size_t)(t0 + 16 + lnLo) * HD;
    v8f accg[2][4], accu[2][4];
#pragma unroll
    for (int s = 0; s < 2; s++)
#pragma unroll
        for (int j = 0; j < 4; j++) {
            accg[s][j] = v8f{0.f,0.f,0.f,0.f,0.f,0.f,0.f,0.f};
            accu[s][j] = v8f{0.f,0.f,0.f,0.f,0.f,0.f,0.f,0.f};
        }
    for (int k0 = 0; k0 < HD; k0 += 32) {
        __builtin_prefetch(wsgT + (size_t)(n0 + lnLo) * HD + k0 + 256, 0, 3);
        __builtin_prefetch(wsuT + (size_t)(n0 + lnLo) * HD + k0 + 256, 0, 3);
        v16bf a0 = load_a_frag(arow0 + k0, laneHi);
        v16bf a1 = load_a_frag(arow1 + k0, laneHi);
#pragma unroll
        for (int j = 0; j < 4; j++) {
            v16bf b = load_bt_frag(wsgT, HD, n0 + 16 * j + lnLo, k0, laneHi);
            accg[0][j] = wmma_bf16(a0, b, accg[0][j]);
            accg[1][j] = wmma_bf16(a1, b, accg[1][j]);
        }
#pragma unroll
        for (int j = 0; j < 4; j++) {
            v16bf b = load_bt_frag(wsuT, HD, n0 + 16 * j + lnLo, k0, laneHi);
            accu[0][j] = wmma_bf16(a0, b, accu[0][j]);
            accu[1][j] = wmma_bf16(a1, b, accu[1][j]);
        }
    }
#pragma unroll
    for (int s = 0; s < 2; s++)
#pragma unroll
        for (int j = 0; j < 4; j++) {
            const int n = n0 + 16 * j + lnLo;
#pragma unroll
            for (int r = 0; r < 8; r++) {
                const int t = t0 + s * 16 + laneHi * 8 + r;
                float h = silu(accg[s][j][r]) * accu[s][j][r];
                sbuf[(size_t)t * ISD + n] = f2bf(h);
            }
        }
}

// ---- kernel 6: shared down proj + final combine -> d_out ---------------
// wsdT: [HD][ISD] bf16
__global__ __launch_bounds__(32) void k_shared_down(const unsigned short* __restrict__ wsdT,
                                                    const unsigned short* __restrict__ sbuf,
                                                    const float* __restrict__ out2,
                                                    float* __restrict__ out) {
    const int t0 = blockIdx.y * 32;
    const int n0 = blockIdx.x * 64;
    const int lane = threadIdx.x, laneHi = lane >> 4, lnLo = lane & 15;
    const unsigned short* arow0 = sbuf + (size_t)(t0 + lnLo) * ISD;
    const unsigned short* arow1 = sbuf + (size_t)(t0 + 16 + lnLo) * ISD;
    v8f acc[2][4];
#pragma unroll
    for (int s = 0; s < 2; s++)
#pragma unroll
        for (int j = 0; j < 4; j++) acc[s][j] = v8f{0.f,0.f,0.f,0.f,0.f,0.f,0.f,0.f};
    for (int k0 = 0; k0 < ISD; k0 += 32) {
        __builtin_prefetch(wsdT + (size_t)(n0 + lnLo) * ISD + k0 + 256, 0, 3);
        v16bf a0 = load_a_frag(arow0 + k0, laneHi);
        v16bf a1 = load_a_frag(arow1 + k0, laneHi);
#pragma unroll
        for (int j = 0; j < 4; j++) {
            v16bf b = load_bt_frag(wsdT, ISD, n0 + 16 * j + lnLo, k0, laneHi);
            acc[0][j] = wmma_bf16(a0, b, acc[0][j]);
            acc[1][j] = wmma_bf16(a1, b, acc[1][j]);
        }
    }
#pragma unroll
    for (int s = 0; s < 2; s++)
#pragma unroll
        for (int j = 0; j < 4; j++) {
            const int n = n0 + 16 * j + lnLo;
#pragma unroll
            for (int r = 0; r < 8; r++) {
                const int t = t0 + s * 16 + laneHi * 8 + r;
                float routed = out2[((size_t)t * 2 + 0) * HD + n] +
                               out2[((size_t)t * 2 + 1) * HD + n];
                out[(size_t)t * HD + n] = acc[s][j][r] + routed;
            }
        }
}

// ---- workspace layout --------------------------------------------------
constexpr size_t XB_OFF  = 0;                                       // bf16 x [NT][HD]
constexpr size_t CNT_OFF = XB_OFF + (size_t)NT * HD * 2;            // int [NE]
constexpr size_t ST_OFF  = CNT_OFF + 256;                           // slot_token [NE][CAP]
constexpr size_t SD_OFF  = ST_OFF + (size_t)NE * CAP * 4;           // slot_dst
constexpr size_t SW_OFF  = SD_OFF + (size_t)NE * CAP * 4;           // slot_wt
constexpr size_t HB_OFF  = SW_OFF + (size_t)NE * CAP * 4;           // hbuf bf16 [NE][CAP][ID]
constexpr size_t O2_OFF  = HB_OFF + (size_t)NE * CAP * ID * 2;      // out2 f32 [NT*2][HD]
constexpr size_t SB_OFF  = O2_OFF + (size_t)NT * 2 * HD * 4;        // sbuf bf16 [NT][ISD]
constexpr size_t WTG_OFF = SB_OFF + (size_t)NT * ISD * 2;           // wtg bf16 [NE][ID][HD]
constexpr size_t WTU_OFF = WTG_OFF + (size_t)NE * ID * HD * 2;      // wtu bf16 [NE][ID][HD]
constexpr size_t WTD_OFF = WTU_OFF + (size_t)NE * ID * HD * 2;      // wtd bf16 [NE][HD][ID]
constexpr size_t SGT_OFF = WTD_OFF + (size_t)NE * ID * HD * 2;      // wsgT bf16 [ISD][HD]
constexpr size_t SUT_OFF = SGT_OFF + (size_t)ISD * HD * 2;          // wsuT bf16 [ISD][HD]
constexpr size_t SDT_OFF = SUT_OFF + (size_t)ISD * HD * 2;          // wsdT bf16 [HD][ISD]

extern "C" void kernel_launch(void* const* d_in, const int* in_sizes, int n_in,
                              void* d_out, int out_size, void* d_ws, size_t ws_size,
                              hipStream_t stream) {
    const float* x       = (const float*)d_in[0];
    const float* gate_w  = (const float*)d_in[1];
    const float* w_gate  = (const float*)d_in[2];
    const float* w_up    = (const float*)d_in[3];
    const float* w_down  = (const float*)d_in[4];
    const float* ws_gate = (const float*)d_in[5];
    const float* ws_up   = (const float*)d_in[6];
    const float* ws_down = (const float*)d_in[7];
    float* out = (float*)d_out;

    char* ws = (char*)d_ws;
    unsigned short* xb   = (unsigned short*)(ws + XB_OFF);
    int*   counts        = (int*)(ws + CNT_OFF);
    int*   slot_token    = (int*)(ws + ST_OFF);
    int*   slot_dst      = (int*)(ws + SD_OFF);
    float* slot_wt       = (float*)(ws + SW_OFF);
    unsigned short* hbuf = (unsigned short*)(ws + HB_OFF);
    float* out2          = (float*)(ws + O2_OFF);
    unsigned short* sbuf = (unsigned short*)(ws + SB_OFF);
    unsigned short* wtg  = (unsigned short*)(ws + WTG_OFF);
    unsigned short* wtu  = (unsigned short*)(ws + WTU_OFF);
    unsigned short* wtd  = (unsigned short*)(ws + WTD_OFF);
    unsigned short* wsgT = (unsigned short*)(ws + SGT_OFF);
    unsigned short* wsuT = (unsigned short*)(ws + SUT_OFF);
    unsigned short* wsdT = (unsigned short*)(ws + SDT_OFF);

    hipMemsetAsync(counts, 0, NE * sizeof(int), stream);
    hipMemsetAsync(out2, 0, (size_t)NT * 2 * HD * sizeof(float), stream);

    // ---- prep: bf16-transpose all weights (amortized across token tiles)
    k_transpose_cvt<<<dim3(ID / 32, HD / 32, NE), 256, 0, stream>>>(
        w_gate, wtg, HD, ID, (size_t)HD * ID, (size_t)ID * HD);
    k_transpose_cvt<<<dim3(ID / 32, HD / 32, NE), 256, 0, stream>>>(
        w_up, wtu, HD, ID, (size_t)HD * ID, (size_t)ID * HD);
    k_transpose_cvt<<<dim3(HD / 32, ID / 32, NE), 256, 0, stream>>>(
        w_down, wtd, ID, HD, (size_t)ID * HD, (size_t)HD * ID);
    k_transpose_cvt<<<dim3(ISD / 32, HD / 32, 1), 256, 0, stream>>>(
        ws_gate, wsgT, HD, ISD, 0, 0);
    k_transpose_cvt<<<dim3(ISD / 32, HD / 32, 1), 256, 0, stream>>>(
        ws_up, wsuT, HD, ISD, 0, 0);
    k_transpose_cvt<<<dim3(HD / 32, ISD / 32, 1), 256, 0, stream>>>(
        ws_down, wsdT, ISD, HD, 0, 0);

    // ---- main pipeline
    k_cvt_x<<<(NT * HD) / 1024, 256, 0, stream>>>(x, xb);
    k_router<<<NT, 32, 0, stream>>>(x, gate_w, counts, slot_token, slot_dst, slot_wt);
    k_gateup<<<dim3(ID / 64, CAP / 32, NE), 32, 0, stream>>>(wtg, wtu, xb, counts,
                                                             slot_token, hbuf);
    k_down<<<dim3(HD / 64, CAP / 32, NE), 32, 0, stream>>>(wtd, hbuf, counts,
                                                           slot_dst, slot_wt, out2);
    k_shared_gateup<<<dim3(ISD / 64, NT / 32), 32, 0, stream>>>(wsgT, wsuT, xb, sbuf);
    k_shared_down<<<dim3(HD / 64, NT / 32), 32, 0, stream>>>(wsdT, sbuf, out2, out);
}
